// Triplet_4329327035153
// MI455X (gfx1250) — matrix operand order repaired and checked by reading, not compiled, not run on hardware
//
#include <hip/hip_runtime.h>
#include <stdint.h>

#define DHID 200
#define TSTEPS 4096
#define NC 12
#define NL 11
#define OUTN 512
#define FEATN ((NC + NL) * DHID)   // 4600

#define BLK 512                    // 16 waves per WG
#define UNITS_PER_WG 128           // 4 lanes per hidden unit, 8 units per wave
#define NWG_C 19                   // ceil(2400/128)
#define NWG_L 18                   // ceil(2200/128)
#define NWG_TOT (NWG_C + NWG_L)

// packed-weight geometry: per wave-group (8 units x 4 lanes = 32 lanes),
// each lane owns 6 chunks (ih r/z/n, hh r/z/n) of 50 bf16, padded to 56.
// layout: [group][iter 0..41][lane 0..31][8 bf16]  -> fully coalesced b128.
#define CHUNK_PAD 56
#define ITERS_PER_LANE 42                  // 6*56/8
#define ELEMS_PER_GROUP (ITERS_PER_LANE * 32 * 8)   // 10752 bf16
#define GRP_C (NWG_C * 16)                 // 304 groups (padded)
#define GRP_L (NWG_L * 16)                 // 288 groups (padded)

typedef __attribute__((ext_vector_type(16))) __bf16 v16bf;
typedef __attribute__((ext_vector_type(8)))  float  v8f;

__device__ __forceinline__ float bf_lo(unsigned w) { return __uint_as_float(w << 16); }
__device__ __forceinline__ float bf_hi(unsigned w) { return __uint_as_float(w & 0xFFFF0000u); }

// ---------------------------------------------------------------------------
// Kernel 0: repack weights f32 -> bf16 in per-wave coalesced order,
//           zero h buffers + barrier words.
// ---------------------------------------------------------------------------
__global__ void setup_kernel(const float* __restrict__ wih_c, const float* __restrict__ whh_c,
                             const float* __restrict__ wih_l, const float* __restrict__ whh_l,
                             unsigned short* __restrict__ pk_c, unsigned short* __restrict__ pk_l,
                             float* __restrict__ h_c, float* __restrict__ h_l,
                             unsigned* __restrict__ bar)
{
  long idx = (long)blockIdx.x * blockDim.x + threadIdx.x;
  long stride = (long)gridDim.x * blockDim.x;
  const long PEC = (long)GRP_C * ELEMS_PER_GROUP;
  const long PEL = (long)GRP_L * ELEMS_PER_GROUP;
  for (long p = idx; p < PEC + PEL; p += stride) {
    int branch = p >= PEC;
    long q = branch ? (p - PEC) : p;
    const float* Wih = branch ? wih_l : wih_c;
    const float* Whh = branch ? whh_l : whh_c;
    unsigned short* dst = branch ? pk_l : pk_c;
    int UT = (branch ? NL : NC) * DHID;
    long g = q / ELEMS_PER_GROUP;
    int r = (int)(q % ELEMS_PER_GROUP);
    int i = r >> 8;               // iter (42)
    int L = (r >> 3) & 31;        // lane
    int s = r & 7;                // slot within 16B
    int j = i * 8 + s;            // 0..335 within lane
    int c = j / CHUNK_PAD, e = j % CHUNK_PAD;   // chunk, elem
    int ulocal = L >> 2, kl = L & 3;
    long u = g * 8 + ulocal;
    float val = 0.f;
    if (u < UT && e < 50) {
      int n = (int)(u / DHID), d = (int)(u % DHID);
      long row = (long)n * 600 + (c % 3) * DHID + d;
      const float* W = (c < 3) ? Wih : Whh;
      val = W[row * DHID + kl * 50 + e];
    }
    ((__bf16*)dst)[q] = (__bf16)val;            // native v_cvt bf16
  }
  for (long i = idx; i < 2 * NC * DHID; i += stride) h_c[i] = 0.f;
  for (long i = idx; i < 2 * NL * DHID; i += stride) h_l[i] = 0.f;
  if (idx < 8) bar[idx] = 0u;
}

// ---------------------------------------------------------------------------
// Kernel 1: persistent recurrent kernel, both branches, device-scope barrier
// ---------------------------------------------------------------------------
__global__ void __launch_bounds__(BLK, 1) rnn_kernel(
    const float* __restrict__ xc, const float* __restrict__ xl,
    const float* __restrict__ dbn_c, const float* __restrict__ dbn_l,
    const float* __restrict__ bih_c, const float* __restrict__ bhh_c,
    const float* __restrict__ bih_l, const float* __restrict__ bhh_l,
    const unsigned short* __restrict__ pk_c, const unsigned short* __restrict__ pk_l,
    float* __restrict__ h_c, float* __restrict__ h_l, unsigned* __restrict__ bar)
{
  // 16 rows = WMMA tile size: rows NN..15 and cols 200..207 stay zero forever,
  // so fragment builds / D-stores need no lane-divergent guards.
  __shared__ float h_lds[16][208];
  __shared__ float msg_lds[16][208];
  __shared__ float xin_lds[2][208];    // cols 200..207 stay zero

  const int tid = threadIdx.x;
  const int wg = blockIdx.x;
  const int branch = (wg >= NWG_C) ? 1 : 0;
  const int gwg = branch ? (wg - NWG_C) : wg;
  const int NN = branch ? NL : NC;
  const int UT = NN * DHID;
  const int NWGB = branch ? NWG_L : NWG_C;
  const int XS = (NN + 2) * DHID;

  const float* x   = branch ? xl : xc;
  const float* dbn = branch ? dbn_l : dbn_c;
  const float* bih = branch ? bih_l : bih_c;
  const float* bhh = branch ? bhh_l : bhh_c;
  const unsigned short* pk = branch ? pk_l : pk_c;
  float* hbuf = branch ? h_l : h_c;    // [2][UT], double-buffered
  unsigned* cnt   = &bar[branch];
  unsigned* phase = &bar[2 + branch];

  const int u0 = gwg * UNITS_PER_WG;
  const int nlo = u0 / DHID;
  int ulast = u0 + UNITS_PER_WG - 1; if (ulast >= UT) ulast = UT - 1;
  const int nhi = ulast / DHID;
  const int nnode = nhi - nlo + 1;     // 1 or 2

  // zero all of h_lds (padding rows/cols persist) + xin padding, once
  for (int i = tid; i < 16 * 208; i += BLK) ((float*)h_lds)[i] = 0.f;
  if (tid < 16) xin_lds[tid >> 3][DHID + (tid & 7)] = 0.f;

  // A fragment for WMMA (dbn, 16x32 bf16, zero-padded), built once
  const int lane = tid & 31;
  union AF { v16bf v; __bf16 b[16]; } afrag;
  {
    int m = lane & 15;
#pragma unroll
    for (int e = 0; e < 16; ++e) {
      int k = e + (e < 8 ? 0 : 8) + (lane < 16 ? 0 : 8);   // 16-bit A layout
      float v = (m < NN && k < NN) ? dbn[m * NN + k] : 0.f;
      afrag.b[e] = (__bf16)v;
    }
  }
  const int wv = tid >> 5;             // wave id 0..15

  // per-unit constants
  const int ul = tid >> 2;             // 0..127
  const int kl = tid & 3;              // 0..3
  const int unit = u0 + ul;
  const bool valid = unit < UT;
  const int un = valid ? (unit / DHID) : nlo;
  const int ud = valid ? (unit % DHID) : 0;
  const int unloc = un - nlo;

  // packed weight base for this lane: [group][iter][lane][16B]
  const uint4* wq = (const uint4*)(const void*)pk
                  + (size_t)(gwg * 16 + wv) * (ITERS_PER_LANE * 32) + lane;

  float b_ir = 0, b_iz = 0, b_in = 0, b_hr = 0, b_hz = 0, b_hn = 0;
  if (valid && kl == 0) {
    b_ir = bih[un * 600 + ud];
    b_iz = bih[un * 600 + DHID + ud];
    b_in = bih[un * 600 + 2 * DHID + ud];
    b_hr = bhh[un * 600 + ud];
    b_hz = bhh[un * 600 + DHID + ud];
    b_hn = bhh[un * 600 + 2 * DHID + ud];
  }
  const unsigned hldsbase = (unsigned)(size_t)(&h_lds[0][0]);
  __syncthreads();

  for (int t = 0; t < TSTEPS; ++t) {
    const float* hrd = hbuf + (size_t)(t & 1) * UT;
    float* hwr = hbuf + (size_t)((t + 1) & 1) * UT;

    // 1. async-copy h(t-1) global -> LDS (ASYNCcnt path, no VGPR round trip)
    for (int i = tid * 4; i < UT; i += BLK * 4) {
      unsigned ldsa = hldsbase + (unsigned)(((i / DHID) * 208 + (i % DHID)) * 4);
      const float* ga = hrd + i;
      asm volatile("global_load_async_to_lds_b128 %0, %1, off"
                   :: "v"(ldsa), "v"(ga) : "memory");
    }
    asm volatile("s_wait_asynccnt 0" ::: "memory");
    __syncthreads();

    // 2. msg = dbn @ h via v_wmma_f32_16x16x32_bf16, 13 column tiles
    if (wv < 13) {
      int d0 = wv * 16;
      union BF { v16bf v; __bf16 b[16]; } bfrag;
      int ncol = lane & 15;
#pragma unroll
      for (int e = 0; e < 16; ++e) {
        // table order: k = e + (e<8?0:8) + (lane<16?0:8); k>=16 are all zero
        // (zero-padded rows cover k>=NN).
        int k = e + (e < 8 ? 0 : 8) + (lane < 16 ? 0 : 8);
        float hv = h_lds[k & 15][d0 + ncol];
        bfrag.b[e] = (__bf16)(k < 16 ? hv : 0.f);
      }
      v8f acc = {0.f, 0.f, 0.f, 0.f, 0.f, 0.f, 0.f, 0.f};
      acc = __builtin_amdgcn_wmma_f32_16x16x32_bf16(false, afrag.v, false, bfrag.v,
                                                    (short)0, acc, false, false);
#pragma unroll
      for (int j = 0; j < 8; ++j) {
        int row = j + (lane < 16 ? 0 : 8);
        msg_lds[row][d0 + ncol] = acc[j];    // unconditional, padded rows unused
      }
    }
    __syncthreads();

    // 3. xin = node_x + msg + cpt for this WG's (<=2) nodes
    const float* xrow = x + (long)t * XS;
    if (tid < nnode * DHID) {
      int nd = tid / DHID, d = tid % DHID;
      int node = nlo + nd;                   // always < NN by construction
      float cpt = xrow[NN * DHID + d] + xrow[(NN + 1) * DHID + d];
      xin_lds[nd][d] = xrow[node * DHID + d] + msg_lds[node][d] + cpt;
    }
    if (t + 1 < TSTEPS && tid < 64)
      __builtin_prefetch(x + (long)(t + 1) * XS + tid * 64, 0, 1);
    __syncthreads();

    // 4. six 50-length partial dots per lane; packed bf16 weights,
    //    42 fully-coalesced b128 loads per thread per step.
    float accs[6];
    {
      const float* xv = &xin_lds[unloc][kl * 50];
      const float* hv = &h_lds[un][kl * 50];
#pragma unroll
      for (int c = 0; c < 6; ++c) {
        const float* vec = (c < 3) ? xv : hv;
        float a = 0.f;
#pragma unroll
        for (int i = 0; i < 7; ++i) {
          uint4 w = wq[(c * 7 + i) * 32];
          const float* vb = vec + i * 8;
          a = fmaf(bf_lo(w.x), vb[0], a); a = fmaf(bf_hi(w.x), vb[1], a);
          a = fmaf(bf_lo(w.y), vb[2], a); a = fmaf(bf_hi(w.y), vb[3], a);
          a = fmaf(bf_lo(w.z), vb[4], a); a = fmaf(bf_hi(w.z), vb[5], a);
          a = fmaf(bf_lo(w.w), vb[6], a); a = fmaf(bf_hi(w.w), vb[7], a);
        }
        accs[c] = a;
      }
    }
#pragma unroll
    for (int c = 0; c < 6; ++c) {
      accs[c] += __shfl_xor(accs[c], 1, 32);
      accs[c] += __shfl_xor(accs[c], 2, 32);
    }

    if (valid && kl == 0) {
      float ir = accs[0] + b_ir, iz = accs[1] + b_iz, inn = accs[2] + b_in;
      float hr = accs[3] + b_hr, hz = accs[4] + b_hz, hn = accs[5] + b_hn;
      float r = 1.f / (1.f + __expf(-(ir + hr)));
      float z = 1.f / (1.f + __expf(-(iz + hz)));
      float ng = tanhf(inn + r * hn);
      float hold = h_lds[un][ud];
      hwr[unit] = (1.f - z) * ng + z * hold;
    }

    // 5. device-scope barrier among this branch's WGs (monotonic counter)
    __builtin_amdgcn_fence(__ATOMIC_RELEASE, "agent");
    __syncthreads();
    if (tid == 0) {
      unsigned ticket = __hip_atomic_fetch_add(cnt, 1u, __ATOMIC_RELAXED,
                                               __HIP_MEMORY_SCOPE_AGENT);
      if ((int)(ticket % (unsigned)NWGB) == NWGB - 1)
        __hip_atomic_fetch_add(phase, 1u, __ATOMIC_RELEASE,
                               __HIP_MEMORY_SCOPE_AGENT);
      while (__hip_atomic_load(phase, __ATOMIC_ACQUIRE,
                               __HIP_MEMORY_SCOPE_AGENT) < (unsigned)(t + 1)) {
        __builtin_amdgcn_s_sleep(1);
      }
    }
    __syncthreads();
    __builtin_amdgcn_fence(__ATOMIC_ACQUIRE, "agent");
  }
}

// ---------------------------------------------------------------------------
// Kernel 2: final FC  out = fc_W @ feat + fc_b   (feat = [h_chart, h_lab])
// ---------------------------------------------------------------------------
__global__ void fc_kernel(const float* __restrict__ fcW, const float* __restrict__ fcb,
                          const float* __restrict__ hc, const float* __restrict__ hl,
                          float* __restrict__ out)
{
  __shared__ float red[256];
  int o = blockIdx.x;
  int tid = threadIdx.x;
  float s = 0.f;
  const float* wr = fcW + (long)o * FEATN;
  for (int j = tid; j < FEATN; j += 256) {
    float f = (j < NC * DHID) ? hc[j] : hl[j - NC * DHID];
    s = fmaf(wr[j], f, s);
  }
  red[tid] = s;
  __syncthreads();
  for (int off = 128; off > 0; off >>= 1) {
    if (tid < off) red[tid] += red[tid + off];
    __syncthreads();
  }
  if (tid == 0) out[o] = red[0] + fcb[o];
}

// ---------------------------------------------------------------------------
extern "C" void kernel_launch(void* const* d_in, const int* in_sizes, int n_in,
                              void* d_out, int out_size, void* d_ws, size_t ws_size,
                              hipStream_t stream)
{
  (void)in_sizes; (void)n_in; (void)out_size; (void)ws_size;
  const float* xc    = (const float*)d_in[0];
  const float* xl    = (const float*)d_in[1];
  const float* dbn_c = (const float*)d_in[2];
  const float* dbn_l = (const float*)d_in[3];
  const float* wih_c = (const float*)d_in[4];
  const float* whh_c = (const float*)d_in[5];
  const float* bih_c = (const float*)d_in[6];
  const float* bhh_c = (const float*)d_in[7];
  const float* wih_l = (const float*)d_in[8];
  const float* whh_l = (const float*)d_in[9];
  const float* bih_l = (const float*)d_in[10];
  const float* bhh_l = (const float*)d_in[11];
  const float* fcW   = (const float*)d_in[12];
  const float* fcb   = (const float*)d_in[13];
  float* out = (float*)d_out;

  char* ws = (char*)d_ws;
  size_t off = 0;
  unsigned* bar = (unsigned*)(ws + off); off += 256;
  float* h_c = (float*)(ws + off); off += (size_t)2 * NC * DHID * 4;
  float* h_l = (float*)(ws + off); off += (size_t)2 * NL * DHID * 4;
  off = (off + 255) & ~(size_t)255;
  unsigned short* pk_c = (unsigned short*)(ws + off); off += (size_t)GRP_C * ELEMS_PER_GROUP * 2;
  unsigned short* pk_l = (unsigned short*)(ws + off); off += (size_t)GRP_L * ELEMS_PER_GROUP * 2;

  setup_kernel<<<2048, 256, 0, stream>>>(wih_c, whh_c, wih_l, whh_l,
                                         pk_c, pk_l, h_c, h_l, bar);
  rnn_kernel<<<NWG_TOT, BLK, 0, stream>>>(xc, xl, dbn_c, dbn_l,
                                          bih_c, bhh_c, bih_l, bhh_l,
                                          pk_c, pk_l, h_c, h_l, bar);
  // final h is in parity-0 buffers (T=4096 even: last step t=4095 writes buf 0)
  fc_kernel<<<OUTN, 256, 0, stream>>>(fcW, fcb, h_c, h_l, out);
}
